// DCNv2_687194767452
// MI455X (gfx1250) — matrix operand order repaired and checked by reading, not compiled, not run on hardware
//
#include <hip/hip_runtime.h>
#include <math.h>

typedef __attribute__((ext_vector_type(16))) _Float16 v16h;
typedef __attribute__((ext_vector_type(8)))  _Float16 v8h;
typedef __attribute__((ext_vector_type(4)))  _Float16 v4h;
typedef __attribute__((ext_vector_type(8)))  float    v8f;
typedef __attribute__((ext_vector_type(4)))  float    f4;

#define HX 96
#define WX 96
#define CX 256
#define FX 256
#define MTOT 36864      /* B*H*W */
#define KDIM 2304       /* 9*256 */
#define NCHUNK 72       /* KDIM/32 */

/* workspace layout (bytes) */
#define WS_OMT  0                    /* 32*2304 f16    = 147456 B  */
#define WS_KT   147456               /* 256*2304 f16   = 1179648 B */
#define WS_OM   1327104              /* 36864*32 f32   = 4718592 B */
#define WS_XH   6045696              /* 36864*256 f16  = 18874368 B*/
#define WS_ZERO 24920064             /* 256 B of zeros             */
#define ZOFF    (WS_ZERO - WS_XH)    /* zero page as byte offset from xh */

static __device__ __forceinline__ v16h cat16(v8h lo, v8h hi) {
  return __builtin_shufflevector(lo, hi, 0,1,2,3,4,5,6,7,8,9,10,11,12,13,14,15);
}

/* ---- one-shot weight conversion: f32 -> f16, transposed for B-fragments ---- */
__global__ void prep_weights(const float* __restrict__ omk,
                             const float* __restrict__ kern,
                             _Float16* __restrict__ omT,
                             _Float16* __restrict__ kT) {
  int idx = blockIdx.x * 256 + threadIdx.x;
  const int total1 = 32 * KDIM;
  const int total2 = FX * KDIM;
  if (idx < total1) {
    int oc = idx / KDIM, k = idx % KDIM;
    float v = (oc < 27) ? omk[k * 27 + oc] : 0.f;
    omT[idx] = (_Float16)v;
  } else if (idx < total1 + total2) {
    int j = idx - total1;
    int n = j / KDIM, k = j % KDIM;
    kT[j] = (_Float16)kern[k * FX + n];
  }
}

/* ---- x: f32 -> f16 (feeds the async im2col copies), + zero page ---- */
__global__ void prep_x(const float* __restrict__ x, _Float16* __restrict__ xh,
                       _Float16* __restrict__ zero) {
  int t = blockIdx.x * 256 + threadIdx.x;
  f4 v = *(const f4*)(x + (size_t)t * 4);
  v4h h;
  h.x = (_Float16)v.x; h.y = (_Float16)v.y;
  h.z = (_Float16)v.z; h.w = (_Float16)v.w;
  *(v4h*)(xh + (size_t)t * 4) = h;
  if (blockIdx.x == 0 && threadIdx.x < 32)
    ((unsigned long long*)zero)[threadIdx.x] = 0ull;
}

/* ---- offset/mask conv as GEMM [36864x2304]x[2304x32], async im2col ---- */
__global__ void __launch_bounds__(256)
om_gemm(const _Float16* __restrict__ xh, const _Float16* __restrict__ omT,
        const float* __restrict__ om_bias, float* __restrict__ omOut) {
  __shared__ _Float16 At[2][64][48];  /* 48-half stride keeps b128 aligned */
  int tid  = threadIdx.x;
  int wave = tid >> 5, lane = tid & 31;
  int mtile = wave & 3, ntile = wave >> 2;   /* 4 m-tiles x 2 n-tiles */
  int wgM = blockIdx.x * 64;

  /* fixed per-thread copy slot: 16 B (8 channels) of one tile row */
  int crow = tid >> 2, seg = tid & 3;
  int m = wgM + crow;
  int b = m / 9216, pix = m % 9216;
  int ho = pix / WX, wo = pix % WX;
  int hom1 = ho - 1, wom1 = wo - 1;          /* top-left of 3x3 window */
  int base0 = b * 9216;
  const char* xb = (const char*)xh;
  unsigned lds0 = (unsigned)(unsigned long long)&At[0][crow][seg * 8];

  /* prologue: async-copy chunk 0 (kh=0,kw=0,cb=0) into buffer 0 */
  {
    bool valid = (hom1 >= 0) & (wom1 >= 0);
    int pix2 = base0 + hom1 * WX + wom1;
    int off = valid ? ((pix2 << 9) + seg * 16) : ZOFF;   /* branchless */
    const char* src = xb + off;
    asm volatile("global_load_async_to_lds_b128 %0, %1, off"
                 :: "v"(lds0), "v"(src) : "memory");
  }

  v8f acc = {};
  for (int ch = 0; ch < NCHUNK; ++ch) {
    int p = ch & 1;
    asm volatile("s_wait_asynccnt 0" ::: "memory"); /* buffer p landed */
    __syncthreads();                                /* all copies done,
                                                       all reads of p^1 done */
    /* issue next chunk into the other buffer (redundant reissue on last) */
    {
      int chn = (ch + 1 < NCHUNK) ? ch + 1 : NCHUNK - 1;
      int kh = chn / 24, kw = (chn % 24) / 8, cb = chn & 7;
      int y = hom1 + kh, xx = wom1 + kw;
      bool valid = (y >= 0) & (y < HX) & (xx >= 0) & (xx < WX);
      int pix2 = base0 + y * WX + xx;
      int off = valid ? ((pix2 << 9) + (cb << 6) + seg * 16) : ZOFF;
      const char* src = xb + off;
      unsigned lds = lds0 + (unsigned)((p ^ 1) * (64 * 48 * 2));
      asm volatile("global_load_async_to_lds_b128 %0, %1, off"
                   :: "v"(lds), "v"(src) : "memory");
    }
    /* consume buffer p */
    int arow = mtile * 16 + (lane & 15);
    int ak0  = (lane >> 4) * 16;
    v8h alo = *(const v8h*)&At[p][arow][ak0];
    v8h ahi = *(const v8h*)&At[p][arow][ak0 + 8];
    v16h a = cat16(alo, ahi);
    int bcol = ntile * 16 + (lane & 15);
    const _Float16* bp = omT + bcol * KDIM + ch * 32 + ((lane >> 4) << 4);
    v16h bf = cat16(*(const v8h*)bp, *(const v8h*)(bp + 8));
    acc = __builtin_amdgcn_wmma_f32_16x16x32_f16(false, a, false, bf,
                                                 (short)0, acc, false, false);
  }
  int gn = ntile * 16 + (lane & 15);
  float bb = (gn < 27) ? om_bias[gn] : 0.f;
#pragma unroll
  for (int r = 0; r < 8; ++r) {
    int gm = wgM + mtile * 16 + ((lane >> 4) << 3) + r;
    omOut[(gm << 5) + gn] = acc[r] + bb;
  }
}

/* ---- fused deformable-sample + GEMM: cols tile never leaves LDS ---- */
__global__ void __launch_bounds__(256)
dcn_main(const float* __restrict__ x, const float* __restrict__ omOut,
         const _Float16* __restrict__ kT, const float* __restrict__ bias,
         float* __restrict__ out) {
  __shared__ _Float16 At[2][64][48];
  __shared__ int   sOff[576][4];
  __shared__ float sW[576][4];
  int tid  = threadIdx.x;
  int wave = tid >> 5, lane = tid & 31;
  int mtile = wave & 3, whi = wave >> 2;   /* 4 m-tiles x (2x8) n-tiles */
  int wgM = blockIdx.x * 64;

  /* per-(row,kpoint) bilinear metadata: clamped indices + mask-folded weights */
  for (int idx = tid; idx < 576; idx += 256) {
    int row = idx / 9, kp = idx % 9;
    int m = wgM + row;
    int b = m / 9216, pix = m % 9216;
    int ho = pix / WX, wo = pix % WX;
    const float* omr = omOut + ((size_t)m << 5);
    float dy = omr[2 * kp];
    float dx = omr[2 * kp + 1];
    float msk = 2.f / (1.f + __expf(-omr[18 + kp]));
    float py = (float)(ho - 1 + kp / 3) + dy;
    float px = (float)(wo - 1 + kp % 3) + dx;
    float y0 = floorf(py), x0 = floorf(px);
    float fy = py - y0, fx = px - x0;
    int iy0 = (int)y0, ix0 = (int)x0;
    float wc[4] = {(1.f - fy) * (1.f - fx), (1.f - fy) * fx,
                   fy * (1.f - fx),         fy * fx};
#pragma unroll
    for (int j = 0; j < 4; ++j) {
      int yy = iy0 + (j >> 1), xx2 = ix0 + (j & 1);
      bool valid = (yy >= 0) && (yy < HX) && (xx2 >= 0) && (xx2 < WX);
      int yc = min(max(yy, 0), HX - 1), xc = min(max(xx2, 0), WX - 1);
      sOff[idx][j] = b * 9216 + yc * WX + xc;
      sW[idx][j]   = valid ? wc[j] * msk : 0.f;
    }
  }
  __syncthreads();

  v8f acc[8];
#pragma unroll
  for (int i = 0; i < 8; ++i) acc[i] = (v8f){};

  int it = 0;
  for (int kp = 0; kp < 9; ++kp) {
    for (int cb = 0; cb < 8; ++cb, ++it) {
      int p = it & 1;
      int cbase = cb << 5;
      /* build sampled+modulated A tile: float4 gathers, branchless */
#pragma unroll
      for (int i = 0; i < 2; ++i) {
        int e = (i << 8) + tid;
        int row = e >> 3, q = (e & 7) << 2;
        int mi = row * 9 + kp;
        f4 v = {0.f, 0.f, 0.f, 0.f};
#pragma unroll
        for (int j = 0; j < 4; ++j) {
          float w = sW[mi][j];
          f4 s = *(const f4*)(x + (((size_t)sOff[mi][j]) << 8) + cbase + q);
          v.x += w * s.x; v.y += w * s.y; v.z += w * s.z; v.w += w * s.w;
        }
        v4h h;
        h.x = (_Float16)v.x; h.y = (_Float16)v.y;
        h.z = (_Float16)v.z; h.w = (_Float16)v.w;
        *(v4h*)&At[p][row][q] = h;
      }
      __syncthreads();   /* tile p built; p re-used only after next barrier */
      int arow = mtile * 16 + (lane & 15);
      int ak0  = (lane >> 4) * 16;
      v8h alo = *(const v8h*)&At[p][arow][ak0];
      v8h ahi = *(const v8h*)&At[p][arow][ak0 + 8];
      v16h a = cat16(alo, ahi);
      int kidx = (kp << 8) + cbase + ((lane >> 4) << 4);
      __builtin_prefetch(kT + (whi * 128 + (lane & 15)) * KDIM + kidx + 32, 0, 1);
#pragma unroll
      for (int nt = 0; nt < 8; ++nt) {
        int bcol = (whi * 8 + nt) * 16 + (lane & 15);
        const _Float16* bp = kT + bcol * KDIM + kidx;
        v16h bf = cat16(*(const v8h*)bp, *(const v8h*)(bp + 8));
        acc[nt] = __builtin_amdgcn_wmma_f32_16x16x32_f16(
            false, a, false, bf, (short)0, acc[nt], false, false);
      }
    }
  }
#pragma unroll
  for (int nt = 0; nt < 8; ++nt) {
    int gn = (whi * 8 + nt) * 16 + (lane & 15);
    float bb = bias[gn];
#pragma unroll
    for (int r = 0; r < 8; ++r) {
      int gm = wgM + mtile * 16 + ((lane >> 4) << 3) + r;
      out[gm * FX + gn] = acc[nt][r] + bb;
    }
  }
}

extern "C" void kernel_launch(void* const* d_in, const int* in_sizes, int n_in,
                              void* d_out, int out_size, void* d_ws, size_t ws_size,
                              hipStream_t stream) {
  (void)in_sizes; (void)n_in; (void)out_size; (void)ws_size;
  const float* x    = (const float*)d_in[0];
  const float* omk  = (const float*)d_in[1];
  const float* omb  = (const float*)d_in[2];
  const float* kern = (const float*)d_in[3];
  const float* bias = (const float*)d_in[4];
  float* out = (float*)d_out;
  char* ws = (char*)d_ws;
  _Float16* omT   = (_Float16*)(ws + WS_OMT);
  _Float16* kT    = (_Float16*)(ws + WS_KT);
  float*    omOut = (float*)(ws + WS_OM);
  _Float16* xh    = (_Float16*)(ws + WS_XH);
  _Float16* zbuf  = (_Float16*)(ws + WS_ZERO);

  int totalPrep = 32 * KDIM + FX * KDIM;
  prep_weights<<<(totalPrep + 255) / 256, 256, 0, stream>>>(omk, kern, omT, kT);
  prep_x<<<(MTOT * CX / 4) / 256, 256, 0, stream>>>(x, xh, zbuf);
  om_gemm<<<MTOT / 64, 256, 0, stream>>>(xh, omT, omb, omOut);
  dcn_main<<<MTOT / 64, 256, 0, stream>>>(x, omOut, kT, bias, out);
}
